// Coding_15479062134879
// MI455X (gfx1250) — matrix operand order repaired and checked by reading, not compiled, not run on hardware
//
#include <hip/hip_runtime.h>

typedef __attribute__((ext_vector_type(16))) _Float16 v16h;
typedef __attribute__((ext_vector_type(8)))  _Float16 v8h;
typedef __attribute__((ext_vector_type(8)))  float    v8f;

#define C_DIM      256
#define N_SPATIAL  4096
#define K_CODES    1024
#define TILE_PTS   128     // points per workgroup (8 waves x 16 rows)
#define STAGE_ROWS 32      // codes staged per pipeline stage (2 WMMA tiles)
#define N_STAGES   (K_CODES / STAGE_ROWS)   // 32
#define WT_STRIDE  264     // halves per LDS codebook row (256 + 8 pad -> 528B, 4-bank step)

#define WMMA_F16(Af, Bf, Cf) \
  __builtin_amdgcn_wmma_f32_16x16x32_f16(false, (Af), false, (Bf), (short)0, (Cf), false, false)

#define SHUF16(lo, hi) \
  __builtin_shufflevector((lo), (hi), 0, 1, 2, 3, 4, 5, 6, 7, 8, 9, 10, 11, 12, 13, 14, 15)

// GEMM (f16 WMMA, f32 accum) + rowwise argmax, software-pipelined:
//   prefetch(stage t+1 -> regs) overlaps compute(stage t); LDS double-buffered,
//   one barrier per 32-code stage; two accumulators break the WMMA D->C chain.
template <bool WF16>
__global__ __launch_bounds__(256) void vq_argmax_kernel(
    const float* __restrict__ x, const float* __restrict__ wf32,
    const _Float16* __restrict__ wf16, float* __restrict__ code_out)
{
  __shared__ __align__(16) _Float16 buf[2][STAGE_ROWS * WT_STRIDE];

  const int lane = threadIdx.x & 31;
  const int wave = threadIdx.x >> 5;
  const int row  = lane & 15;   // A row within wave tile / B column within code tile
  const int h    = lane >> 4;   // lane half selects K sub-range per ISA layout

  // ---- staging role: thread -> (code row in stage, 32-half chunk) ----
  const int sr = threadIdx.x >> 3;   // 0..31
  const int sj = threadIdx.x & 7;    // 0..7

  const int m0 = blockIdx.x * TILE_PTS + wave * 16;     // global point base (wave)
  const int b  = m0 >> 12;                              // /4096 (128 | 4096)
  const int n  = (m0 & (N_SPATIAL - 1)) + row;
  const float* xb = x + (size_t)b * C_DIM * N_SPATIAL + n;

  // ---- Preload A: 16 points x 256 channels as 8 f16 fragments (16x32 each) ----
  // ISA 16-bit A layout: lanes 0-15 hold K {0..7, 16..23}; lanes 16-31 hold {8..15, 24..31}.
  v16h A[8];
  #pragma unroll
  for (int s = 0; s < 8; ++s) {
    const int kc = s * 32;
    #pragma unroll
    for (int j = 0; j < 8; ++j) {
      A[s][j]     = (_Float16)xb[(size_t)(kc + 8 * h + j)      * N_SPATIAL];
      A[s][8 + j] = (_Float16)xb[(size_t)(kc + 16 + 8 * h + j) * N_SPATIAL];
    }
  }

  float bestv[8];
  int   besti[8];
  #pragma unroll
  for (int v = 0; v < 8; ++v) { bestv[v] = -3.4e38f; besti[v] = 0; }

  // ---- prefetch registers (one path is compile-time dead) ----
  v8h   pfh[4];
  float pff[32];

  // Prologue: prefetch stage 0.
  {
    const size_t off = ((size_t)sr << 8) + sj * 32;
    if constexpr (WF16) {
      const v8h* src = (const v8h*)(wf16 + off);
      #pragma unroll
      for (int q = 0; q < 4; ++q) pfh[q] = src[q];
    } else {
      const float* src = wf32 + off;
      #pragma unroll
      for (int q = 0; q < 32; ++q) pff[q] = src[q];
    }
  }

  for (int st = 0; st < N_STAGES; ++st) {
    const int p = st & 1;

    // ---- commit prefetched stage into LDS buffer p ----
    {
      _Float16* dst = &buf[p][sr * WT_STRIDE + sj * 32];
      if constexpr (WF16) {
        v8h* d = (v8h*)dst;
        #pragma unroll
        for (int q = 0; q < 4; ++q) d[q] = pfh[q];
      } else {
        #pragma unroll
        for (int q = 0; q < 32; ++q) dst[q] = (_Float16)pff[q];
      }
    }
    __syncthreads();   // buf[p] visible; all waves done reading buf[p] from st-2

    // ---- issue global prefetch for stage st+1 (hidden behind compute) ----
    if (st + 1 < N_STAGES) {
      const size_t off = ((size_t)((st + 1) * STAGE_ROWS + sr) << 8) + sj * 32;
      if constexpr (WF16) {
        const v8h* src = (const v8h*)(wf16 + off);
        #pragma unroll
        for (int q = 0; q < 4; ++q) pfh[q] = src[q];
      } else {
        const float* src = wf32 + off;
        #pragma unroll
        for (int q = 0; q < 32; ++q) pff[q] = src[q];
      }
    }

    // ---- compute: 2 code tiles x (8 WMMAs over K=256), dual accumulators ----
    #pragma unroll
    for (int u = 0; u < 2; ++u) {
      const _Float16* brow = &buf[p][(u * 16 + row) * WT_STRIDE];
      v8f acc0 = {}, acc1 = {};
      #pragma unroll
      for (int s = 0; s < 8; s += 2) {
        const v8h* wp0 = (const v8h*)(brow + s * 32 + 8 * h);
        const v8h* wp1 = (const v8h*)(brow + (s + 1) * 32 + 8 * h);
        const v16h B0 = SHUF16(wp0[0], wp0[2]);
        const v16h B1 = SHUF16(wp1[0], wp1[2]);
        acc0 = WMMA_F16(A[s],     B0, acc0);
        acc1 = WMMA_F16(A[s + 1], B1, acc1);
      }
      const int kcode = st * STAGE_ROWS + u * 16 + row;   // D column for this lane
      #pragma unroll
      for (int v = 0; v < 8; ++v) {
        const float cv = acc0[v] + acc1[v];
        const bool better = cv > bestv[v];
        besti[v] = better ? kcode : besti[v];
        bestv[v] = better ? cv    : bestv[v];
      }
    }
  }

  // ---- Reduce across the 16 code-slots within each lane half; tie -> lowest k ----
  #pragma unroll
  for (int v = 0; v < 8; ++v) {
    float bv = bestv[v];
    int   bi = besti[v];
    #pragma unroll
    for (int off = 1; off < 16; off <<= 1) {
      const float ov = __shfl_xor(bv, off, 32);
      const int   oi = __shfl_xor(bi, off, 32);
      if (ov > bv || (ov == bv && oi < bi)) { bv = ov; bi = oi; }
    }
    if (row == 0) {
      // D rows: lanes 0-15 -> M = v, lanes 16-31 -> M = v + 8
      code_out[m0 + v + 8 * h] = (float)bi;
    }
  }
}

// One-time codebook f32 -> f16 conversion into workspace.
__global__ __launch_bounds__(256) void w_to_f16_kernel(
    const float* __restrict__ w, _Float16* __restrict__ w16)
{
  const int i = blockIdx.x * 256 + threadIdx.x;   // < 1024*256
  w16[i] = (_Float16)w[i];
}

// x_new[b,c,n] = weight[code[b,n], c]; coalesced writes, codebook gathers hit L2.
__global__ __launch_bounds__(256) void vq_gather_kernel(
    const float* __restrict__ w, const float* __restrict__ codef,
    float* __restrict__ out)
{
  const int idx = blockIdx.x * 256 + threadIdx.x;   // < 16*256*4096
  const int n = idx & (N_SPATIAL - 1);
  const int c = (idx >> 12) & (C_DIM - 1);
  const int b = idx >> 20;
  const int code = (int)codef[(b << 12) + n];
  out[idx] = w[(size_t)code * C_DIM + c];
}

extern "C" void kernel_launch(void* const* d_in, const int* in_sizes, int n_in,
                              void* d_out, int out_size, void* d_ws, size_t ws_size,
                              hipStream_t stream) {
  const float* x = (const float*)d_in[0];   // [16, 256, 64, 64] f32
  const float* w = (const float*)d_in[1];   // [1024, 256] f32 (unit-norm rows)
  float* out = (float*)d_out;               // [65536 code floats][16*256*4096 x_new]

  const int n_points = 16 * N_SPATIAL;                 // 65536
  const int n_elems  = 16 * C_DIM * N_SPATIAL;         // 16777216
  const size_t w16_bytes = (size_t)K_CODES * C_DIM * sizeof(_Float16);

  if (ws_size >= w16_bytes) {
    _Float16* w16 = (_Float16*)d_ws;
    w_to_f16_kernel<<<K_CODES * C_DIM / 256, 256, 0, stream>>>(w, w16);
    vq_argmax_kernel<true><<<n_points / TILE_PTS, 256, 0, stream>>>(x, w, w16, out);
  } else {
    vq_argmax_kernel<false><<<n_points / TILE_PTS, 256, 0, stream>>>(x, w, nullptr, out);
  }
  vq_gather_kernel<<<n_elems / 256, 256, 0, stream>>>(w, out, out + n_points);
}
